// PredRNN_model_31250182046024
// MI455X (gfx1250) — matrix operand order repaired and checked by reading, not compiled, not run on hardware
//
#include <hip/hip_runtime.h>

// ---------------- problem constants ----------------
#define BB   16
#define TT   20
#define TINN 10
#define HWD  16
#define FCC  16
#define NHC  128
#define NLY  4
#define MMR  4096   // B*HW*HW  (GEMM M dimension)

typedef unsigned int   u32;
typedef unsigned short u16;
typedef __attribute__((ext_vector_type(16))) __bf16 bf16x16;
typedef __attribute__((ext_vector_type(8)))  float  f32x8;
typedef __attribute__((ext_vector_type(4)))  u32    u32x4;

union FragAB { bf16x16 v; u32x4 q[2]; };

__device__ __forceinline__ u16 f2bf(float f) {
    u32 u = __float_as_uint(f);
    u32 r = (u + 0x7FFFu + ((u >> 16) & 1u)) >> 16;   // round-to-nearest-even
    return (u16)r;
}
__device__ __forceinline__ float sigm(float x) { return 1.0f / (1.0f + __expf(-x)); }

// ---------------- zero fill ----------------
__global__ void zero_kernel(u32* __restrict__ p, size_t n) {
    size_t i = (size_t)blockIdx.x * blockDim.x + threadIdx.x;
    size_t stride = (size_t)gridDim.x * blockDim.x;
    for (; i < n; i += stride) p[i] = 0u;
}

// ---------------- weight reorder: fp32 [O][Cin][KH][KW] -> bf16 [N=O][Kpad], k=(tap)*Cin+c ----------------
__global__ void reorder_w(const float* __restrict__ W, u16* __restrict__ Wt,
                          int O, int Cin, int cshift, int KHW, int Kreal, int Kpad) {
    int idx = blockIdx.x * blockDim.x + threadIdx.x;
    if (idx >= O * Kpad) return;
    int o = idx / Kpad;
    int k = idx - o * Kpad;
    float v = 0.0f;
    if (k < Kreal) {
        int tap = k >> cshift;
        int c   = k & (Cin - 1);
        v = W[((size_t)(o * Cin + c)) * KHW + tap];
    }
    Wt[idx] = f2bf(v);
}

// ---------------- implicit-GEMM conv via WMMA bf16 (all shapes compile-time) ----------------
// in : bf16 NHWC [B,HW,HW,CIN]    wt : bf16 [NN][KPAD]    out : f32 [M][NN]
// 256 threads = 8 waves (wave32); block tile 128(M) x 128(N); K staged 32 at a time,
// double-buffered LDS with register prefetch; padded LDS rows (40 u16) for conflict-free b128 frag loads.
#define LROW 40   // padded row stride in u16 (32 data + 8 pad)

template<int CIN, int KSZ, int PAD, int NN, int KPAD>
__global__ __launch_bounds__(256) void conv_gemm(
    const u16* __restrict__ in, const u16* __restrict__ wt, float* __restrict__ out)
{
    constexpr int KREAL  = KSZ * KSZ * CIN;
    constexpr int NSTAGE = KPAD / 32;

    __shared__ __align__(16) u16 As[2][128 * LROW];
    __shared__ __align__(16) u16 Bs[2][128 * LROW];

    const int tid  = threadIdx.x;
    const int lane = tid & 31;
    const int wid  = tid >> 5;
    const int wm   = wid & 3;        // 4 waves along M, 32 rows each
    const int wn   = wid >> 2;       // 2 waves along N, 64 cols each
    const int hi   = lane >> 4;      // K-half select (ISA A/B layouts)
    const int ln   = lane & 15;

    const int mBase = blockIdx.y * 128;
    const int nBase = blockIdx.x * 128;

    // staging-loader mapping: thread -> (row 0..127, 16-wide K segment)
    const int lrow = tid >> 1;
    const int lseg = (tid & 1) << 4;
    const int grow = mBase + lrow;
    const int gb   = grow >> 8;
    const int gp   = grow & 255;
    const int gy   = gp >> 4;
    const int gx   = gp & 15;
    const int bn   = nBase + lrow;

    // ---- global fetch helpers (all shape math folds to constants) ----
    auto loadA = [&](int k0, u32x4& z0, u32x4& z1) {
        z0 = (u32x4){0,0,0,0}; z1 = (u32x4){0,0,0,0};
        const int k = k0 + lseg;
        if (KPAD == KREAL || k < KREAL) {
            const int tap = k / CIN;           // CIN is a power of two -> shift
            const int c0  = k % CIN;
            const int ky  = tap / KSZ;         // compile-time magic multiply
            const int kx  = tap % KSZ;
            const int ys  = gy + ky - PAD;
            const int xs  = gx + kx - PAD;
            if (KSZ == 1 || ((unsigned)ys < (unsigned)HWD && (unsigned)xs < (unsigned)HWD)) {
                const u32x4* src = (const u32x4*)(in + ((size_t)((gb * HWD + ys) * HWD + xs)) * CIN + c0);
                z0 = src[0]; z1 = src[1];
            }
        }
    };
    auto loadB = [&](int k0, u32x4& z0, u32x4& z1) {
        z0 = (u32x4){0,0,0,0}; z1 = (u32x4){0,0,0,0};
        if ((NN % 128 == 0) || bn < NN) {
            const u32x4* src = (const u32x4*)(wt + (size_t)bn * KPAD + k0 + lseg);
            z0 = src[0]; z1 = src[1];
        }
    };
    auto stageStore = [&](int buf, const u32x4& a0, const u32x4& a1,
                          const u32x4& b0, const u32x4& b1) {
        u32x4* da = (u32x4*)(As[buf] + lrow * LROW + lseg);
        da[0] = a0; da[1] = a1;
        u32x4* db = (u32x4*)(Bs[buf] + lrow * LROW + lseg);
        db[0] = b0; db[1] = b1;
    };

    f32x8 acc[2][4];
    const f32x8 z8 = {0.f,0.f,0.f,0.f,0.f,0.f,0.f,0.f};
#pragma unroll
    for (int i = 0; i < 2; ++i)
#pragma unroll
        for (int j = 0; j < 4; ++j) acc[i][j] = z8;

    // ---- prologue: stage 0 ----
    {
        u32x4 a0, a1, b0, b1;
        loadA(0, a0, a1);
        loadB(0, b0, b1);
        stageStore(0, a0, a1, b0, b1);
    }
    __syncthreads();

    for (int s = 0; s < NSTAGE; ++s) {
        const int buf = s & 1;
        // prefetch next stage from global into registers (overlaps with WMMA below)
        u32x4 na0, na1, nb0, nb1;
        if (s + 1 < NSTAGE) {
            loadA((s + 1) * 32, na0, na1);
            loadB((s + 1) * 32, nb0, nb1);
        }

        // ---- fragments + 8 WMMAs from current buffer ----
        FragAB a[2];
#pragma unroll
        for (int mi = 0; mi < 2; ++mi) {
            const u32x4* Ar = (const u32x4*)(As[buf] + (wm * 32 + mi * 16 + ln) * LROW);
            a[mi].q[0] = Ar[hi];       // K 0..7   / 8..15
            a[mi].q[1] = Ar[2 + hi];   // K 16..23 / 24..31
        }
#pragma unroll
        for (int ni = 0; ni < 4; ++ni) {
            FragAB bfr;
            const u32x4* Br = (const u32x4*)(Bs[buf] + (wn * 64 + ni * 16 + ln) * LROW);
            bfr.q[0] = Br[hi];
            bfr.q[1] = Br[2 + hi];
#pragma unroll
            for (int mi = 0; mi < 2; ++mi) {
                acc[mi][ni] = __builtin_amdgcn_wmma_f32_16x16x32_bf16(
                    false, a[mi].v, false, bfr.v, (short)0, acc[mi][ni], false, false);
            }
        }

        // store next stage into the other buffer (disjoint from current reads)
        if (s + 1 < NSTAGE) stageStore(buf ^ 1, na0, na1, nb0, nb1);
        __syncthreads();
    }

    // ---- epilogue: C layout lane%16 = N, vgpr v -> M = v + 8*(lane>>4) ----
#pragma unroll
    for (int ni = 0; ni < 4; ++ni) {
        const int n = nBase + wn * 64 + ni * 16 + ln;
        if ((NN % 128 == 0) || n < NN) {
#pragma unroll
            for (int mi = 0; mi < 2; ++mi) {
                const int rbase = mBase + wm * 32 + mi * 16 + hi * 8;
#pragma unroll
                for (int v = 0; v < 8; ++v)
                    out[(size_t)(rbase + v) * NN + n] = acc[mi][ni][v];
            }
        }
    }
}

// ---------------- ST-LSTM gates: c,m update + bf16 copies ----------------
__global__ void gates_kernel(const float* __restrict__ xc, const float* __restrict__ hc,
                             const float* __restrict__ mc,
                             float* c_state, float* m_state,
                             u16* __restrict__ mem_bf, u16* __restrict__ m_bf)
{
    int i = blockIdx.x * blockDim.x + threadIdx.x;
    if (i >= MMR * NHC) return;
    int r = i >> 7, ch = i & 127;
    const float* x = xc + (size_t)r * 896;
    const float* h = hc + (size_t)r * 512;
    const float* m = mc + (size_t)r * 384;

    float it = sigm(x[ch]       + h[ch]);
    float ft = sigm(x[128 + ch] + h[128 + ch] + 1.0f);
    float gt = tanhf(x[256 + ch] + h[256 + ch]);
    float cn = ft * c_state[i] + it * gt;

    float ip = sigm(x[384 + ch] + m[ch]);
    float fp = sigm(x[512 + ch] + m[128 + ch] + 1.0f);
    float gp = tanhf(x[640 + ch] + m[256 + ch]);
    float mn = fp * m_state[i] + ip * gp;

    c_state[i] = cn;
    m_state[i] = mn;
    mem_bf[(size_t)r * 256 + ch]       = f2bf(cn);
    mem_bf[(size_t)r * 256 + 128 + ch] = f2bf(mn);
    m_bf[i] = f2bf(mn);
}

// ---------------- output gate: h = sigmoid(o_x+o_h+oc) * tanh(cc) ----------------
__global__ void outgate_kernel(const float* __restrict__ xc, const float* __restrict__ hc,
                               const float* __restrict__ oc, const float* __restrict__ cc,
                               u16* __restrict__ h_bf)
{
    int i = blockIdx.x * blockDim.x + threadIdx.x;
    if (i >= MMR * NHC) return;
    int r = i >> 7, ch = i & 127;
    float ot = sigm(xc[(size_t)r * 896 + 768 + ch] + hc[(size_t)r * 512 + 384 + ch] + oc[i]);
    h_bf[i] = f2bf(ot * tanhf(cc[i]));
}

// ---------------- input mixing: net = mk*frame + (1-mk)*x_gen  (bf16 NHWC) ----------------
__global__ void netprep_kernel(const float* __restrict__ frames, const float* __restrict__ mask,
                               const float* __restrict__ xgen, u16* __restrict__ net, int t)
{
    int i = blockIdx.x * blockDim.x + threadIdx.x;
    if (i >= MMR * FCC) return;
    int r = i >> 4, c = i & 15;
    int b = r >> 8, p = r & 255;
    float fr = frames[((size_t)(b * TT + t) * 256 + p) * 16 + c];
    float v;
    if (t < TINN) {
        v = fr;
    } else {
        float mk = mask[((size_t)(b * (TT - TINN) + (t - TINN)) * 256 + p) * 16 + c];
        v = mk * fr + (1.0f - mk) * xgen[i];
    }
    net[i] = f2bf(v);
}

// ---------------- scatter x_gen into d_out [B,T,H,W,C] ----------------
__global__ void writeout_kernel(const float* __restrict__ xgen, float* __restrict__ out, int t)
{
    int i = blockIdx.x * blockDim.x + threadIdx.x;
    if (i >= MMR * FCC) return;
    int r = i >> 4, c = i & 15;
    int b = r >> 8, p = r & 255;
    out[((size_t)(b * TT + t) * 256 + p) * 16 + c] = xgen[i];
}

// ---------------- MSE loss reduction ----------------
__global__ void loss_kernel(const float* __restrict__ pred, const float* __restrict__ frames,
                            float* __restrict__ loss)
{
    __shared__ float red[1024];
    const int n = BB * TT * 256 * FCC;   // 1,310,720
    float s = 0.0f;
    for (int i = threadIdx.x; i < n; i += 1024) {
        float d = pred[i] - frames[i];
        s += d * d;
    }
    red[threadIdx.x] = s;
    __syncthreads();
    for (int w = 512; w > 0; w >>= 1) {
        if ((int)threadIdx.x < w) red[threadIdx.x] += red[threadIdx.x + w];
        __syncthreads();
    }
    if (threadIdx.x == 0) loss[0] = red[0] / (float)n;
}

// ---------------- host orchestration ----------------
extern "C" void kernel_launch(void* const* d_in, const int* in_sizes, int n_in,
                              void* d_out, int out_size, void* d_ws, size_t ws_size,
                              hipStream_t stream)
{
    (void)in_sizes; (void)n_in; (void)out_size; (void)ws_size;

    const float* frames = (const float*)d_in[0];
    const float* mask   = (const float*)d_in[1];
    const float* Wx[NLY]; const float* Wh[NLY]; const float* Wm[NLY];
    const float* Wo[NLY]; const float* Wc[NLY];
    for (int l = 0; l < NLY; ++l) {
        Wx[l] = (const float*)d_in[2 + 5 * l + 0];
        Wh[l] = (const float*)d_in[2 + 5 * l + 1];
        Wm[l] = (const float*)d_in[2 + 5 * l + 2];
        Wo[l] = (const float*)d_in[2 + 5 * l + 3];
        Wc[l] = (const float*)d_in[2 + 5 * l + 4];
    }
    const float* Wlast = (const float*)d_in[22];
    float* out = (float*)d_out;

    // ---- bump allocator over workspace ----
    char* base = (char*)d_ws;
    size_t cur = 0;
    auto alloc = [&](size_t bytes) -> char* {
        cur = (cur + 255) & ~(size_t)255;
        char* p = base + cur;
        cur += bytes;
        return p;
    };

    const int KpadX[NLY] = {416, 3200, 3200, 3200};
    const int CinX[NLY]  = {FCC, NHC, NHC, NHC};
    const int CshX[NLY]  = {4, 7, 7, 7};

    u16* wxb[NLY]; u16* whb[NLY]; u16* wmb[NLY]; u16* wob[NLY]; u16* wcb[NLY];
    for (int l = 0; l < NLY; ++l) {
        wxb[l] = (u16*)alloc((size_t)KpadX[l] * 896 * 2);
        whb[l] = (u16*)alloc((size_t)3200 * 512 * 2);
        wmb[l] = (u16*)alloc((size_t)3200 * 384 * 2);
        wob[l] = (u16*)alloc((size_t)6400 * 128 * 2);
        wcb[l] = (u16*)alloc((size_t)256 * 128 * 2);
    }
    u16* wlb = (u16*)alloc((size_t)128 * 16 * 2);

    // state region (must be zero at start of every call)
    size_t stateStart = (cur + 255) & ~(size_t)255;
    u16*   h_bf  = (u16*)alloc((size_t)NLY * MMR * NHC * 2);
    u16*   m_bf  = (u16*)alloc((size_t)MMR * NHC * 2);
    float* c_st  = (float*)alloc((size_t)NLY * MMR * NHC * 4);
    float* m_st  = (float*)alloc((size_t)MMR * NHC * 4);
    float* xgen  = (float*)alloc((size_t)MMR * FCC * 4);
    size_t stateEnd = cur;

    u16*   net_bf = (u16*)alloc((size_t)MMR * FCC * 2);
    u16*   mem_bf = (u16*)alloc((size_t)MMR * 256 * 2);
    float* xc = (float*)alloc((size_t)MMR * 896 * 4);
    float* hc = (float*)alloc((size_t)MMR * 512 * 4);
    float* mc = (float*)alloc((size_t)MMR * 384 * 4);
    float* oc = (float*)alloc((size_t)MMR * 128 * 4);
    float* cc = (float*)alloc((size_t)MMR * 128 * 4);

    // ---- zero the recurrent state ----
    {
        size_t words = (stateEnd - stateStart + 3) / 4;
        zero_kernel<<<2048, 256, 0, stream>>>((u32*)(base + stateStart), words);
    }

    // ---- reorder all weights to bf16 [N][Kpad] ----
    auto reord = [&](const float* W, u16* Wt, int O, int Cin, int csh, int KHW, int Kpad) {
        int total = O * Kpad;
        reorder_w<<<(total + 255) / 256, 256, 0, stream>>>(W, Wt, O, Cin, csh, KHW, KHW * Cin, Kpad);
    };
    for (int l = 0; l < NLY; ++l) {
        reord(Wx[l], wxb[l], 896, CinX[l], CshX[l], 25, KpadX[l]);
        reord(Wh[l], whb[l], 512, NHC, 7, 25, 3200);
        reord(Wm[l], wmb[l], 384, NHC, 7, 25, 3200);
        reord(Wo[l], wob[l], 128, 256, 8, 25, 6400);
        reord(Wc[l], wcb[l], 128, 256, 8, 1, 256);
    }
    reord(Wlast, wlb, 16, NHC, 7, 1, 128);

    const dim3 blk(256);
    const int EW = MMR * NHC;            // 524288 elementwise lanes

    // ---- time loop ----
    for (int t = 0; t < TT; ++t) {
        netprep_kernel<<<(MMR * FCC + 255) / 256, blk, 0, stream>>>(frames, mask, xgen, net_bf, t);

        for (int l = 0; l < NLY; ++l) {
            const u16* inp = (l == 0) ? net_bf : (h_bf + (size_t)(l - 1) * MMR * NHC);
            // xc = conv5x5(x, Wx) -> 896 channels
            if (l == 0)
                conv_gemm<16, 5, 2, 896, 416><<<dim3(7, 32), blk, 0, stream>>>(inp, wxb[l], xc);
            else
                conv_gemm<128, 5, 2, 896, 3200><<<dim3(7, 32), blk, 0, stream>>>(inp, wxb[l], xc);
            // hc = conv5x5(h_prev, Wh) -> 512
            conv_gemm<128, 5, 2, 512, 3200><<<dim3(4, 32), blk, 0, stream>>>(
                h_bf + (size_t)l * MMR * NHC, whb[l], hc);
            // mc = conv5x5(m, Wm) -> 384
            conv_gemm<128, 5, 2, 384, 3200><<<dim3(3, 32), blk, 0, stream>>>(m_bf, wmb[l], mc);
            // gate math -> c_new, m_new, mem=concat(c,m) bf16
            gates_kernel<<<(EW + 255) / 256, blk, 0, stream>>>(
                xc, hc, mc, c_st + (size_t)l * MMR * NHC, m_st, mem_bf, m_bf);
            // oc = conv5x5(mem, Wo) (256->128)
            conv_gemm<256, 5, 2, 128, 6400><<<dim3(1, 32), blk, 0, stream>>>(mem_bf, wob[l], oc);
            // cc = conv1x1(mem, Wc) (256->128)
            conv_gemm<256, 1, 0, 128, 256><<<dim3(1, 32), blk, 0, stream>>>(mem_bf, wcb[l], cc);
            // h = sigmoid(o_x+o_h+oc) * tanh(cc)
            outgate_kernel<<<(EW + 255) / 256, blk, 0, stream>>>(
                xc, hc, oc, cc, h_bf + (size_t)l * MMR * NHC);
        }
        // x_gen = conv1x1(h[3], W_last) (128->16)
        conv_gemm<128, 1, 0, 16, 128><<<dim3(1, 32), blk, 0, stream>>>(
            h_bf + (size_t)3 * MMR * NHC, wlb, xgen);
        writeout_kernel<<<(MMR * FCC + 255) / 256, blk, 0, stream>>>(xgen, out, t);
    }

    // ---- loss = mean((pred - frames)^2) ----
    loss_kernel<<<1, 1024, 0, stream>>>(out, frames, out + (size_t)BB * TT * 256 * FCC);
}
